// Mamba2Block_60567628808786
// MI455X (gfx1250) — matrix operand order repaired and checked by reading, not compiled, not run on hardware
//
#include <hip/hip_runtime.h>

// ---------------- CDNA5 WMMA types ----------------
typedef _Float16 v16h __attribute__((ext_vector_type(16)));
typedef _Float16 v8h  __attribute__((ext_vector_type(8)));
typedef float    v8f  __attribute__((ext_vector_type(8)));
typedef int      v4i  __attribute__((vector_size(16)));   // int32x4 for async b128

union HV { v16h v; v8h h[2]; };

constexpr int cB  = 2;
constexpr int cT  = 4096;
constexpr int cDM = 1024;
constexpr int cDI = 2048;     // d_inner
constexpr int cH  = 32;       // heads
constexpr int cHD = 64;       // headdim
constexpr int cDS = 128;      // d_state
constexpr int cCS = 256;      // chunk
constexpr int cNC = 16;       // chunks per seq
constexpr int cDP = 4384;     // 2*DI + 2*DS + H

// ---- async global->LDS path (gfx1250 GLOBAL_LOAD_ASYNC_TO_LDS_B128) ----
#if defined(__has_builtin)
#if __has_builtin(__builtin_amdgcn_global_load_async_to_lds_b128) && \
    __has_builtin(__builtin_amdgcn_s_wait_asynccnt)
#define USE_ASYNC_LDS 1
#endif
#endif
#ifndef USE_ASYNC_LDS
#define USE_ASYNC_LDS 0
#endif

#if USE_ASYNC_LDS
typedef __attribute__((address_space(1))) v4i* gv4i_p;
typedef __attribute__((address_space(3))) v4i* lv4i_p;
#define GASYNC_B128(gp, lp)                                        \
  __builtin_amdgcn_global_load_async_to_lds_b128(                  \
      (gv4i_p)(gp), (lv4i_p)(lp), 0, 0)
__device__ __forceinline__ void async_wait0() {
  __builtin_amdgcn_s_wait_asynccnt(0);
}
#else
__device__ __forceinline__ void async_wait0() {}
#endif

__device__ __forceinline__ v8f wmma_f16(v16h a, v16h b, v8f c) {
  // v_wmma_f32_16x16x32_f16  D = A*B + C   (wave32, 8-arg form)
  return __builtin_amdgcn_wmma_f32_16x16x32_f16(false, a, false, b,
                                                (short)0, c, false, false);
}
__device__ __forceinline__ v8f vz8() {
  v8f z = {0.f,0.f,0.f,0.f,0.f,0.f,0.f,0.f}; return z;
}

// ---------------- generic f32 -> f16 cast ----------------
__global__ void cast16_kernel(const float* __restrict__ src,
                              _Float16* __restrict__ dst, long n) {
  long i = (long)blockIdx.x * blockDim.x + threadIdx.x;
  if (i < n) dst[i] = (_Float16)src[i];
}

// ---------------- WMMA GEMM: C[M,N] = A[M,K] * B[N,K]^T ----------------
// block = 256 threads = 8 waves; block tile 128(M) x 128(N); wave tile 32x64.
// Double-buffered LDS; async global->LDS copies when available.
__global__ __launch_bounds__(256) void gemm_f16_kernel(
    const _Float16* __restrict__ A, const _Float16* __restrict__ Bm,
    float* __restrict__ C, int M, int N, int K) {
  __shared__ __align__(32) _Float16 As[2][128 * 32];
  __shared__ __align__(32) _Float16 Bsh[2][128 * 32];

  const int tid  = threadIdx.x;
  const int lane = tid & 31;
  const int wave = tid >> 5;
  const int hi   = lane >> 4;     // half-wave select
  const int lm   = lane & 15;
  const int wm   = wave >> 1;     // 0..3  (M)
  const int wn   = wave & 1;      // 0..1  (N)
  const int mBase = blockIdx.y * 128;
  const int nBase = blockIdx.x * 128;

  // staging coords: 128 rows x 32 cols, 2x b128 per thread per matrix
  const int srow = tid >> 1;
  const int scb  = (tid & 1) * 16;

  v8f acc[2][4];
  for (int a = 0; a < 2; ++a) for (int b = 0; b < 4; ++b) acc[a][b] = vz8();

  // ---- stage one 32-wide K tile into buffer `buf` ----
  auto stage = [&](int k0, int buf) {
    const _Float16* ga = A + (long)(mBase + srow) * K + k0 + scb;
    _Float16* la = &As[buf][srow * 32 + scb];
#if USE_ASYNC_LDS
    GASYNC_B128(ga, la);
    GASYNC_B128(ga + 8, la + 8);
#else
    *(v8h*)(la)     = *(const v8h*)(ga);
    *(v8h*)(la + 8) = *(const v8h*)(ga + 8);
#endif
    _Float16* lb = &Bsh[buf][srow * 32 + scb];
    if (nBase + srow < N) {
      const _Float16* gb = Bm + (long)(nBase + srow) * K + k0 + scb;
#if USE_ASYNC_LDS
      GASYNC_B128(gb, lb);
      GASYNC_B128(gb + 8, lb + 8);
#else
      *(v8h*)(lb)     = *(const v8h*)(gb);
      *(v8h*)(lb + 8) = *(const v8h*)(gb + 8);
      __builtin_prefetch(gb + 32, 0, 1);
#endif
    } else {
      v8h z = {};
      *(v8h*)(lb) = z; *(v8h*)(lb + 8) = z;
    }
  };

  int buf = 0;
  stage(0, buf);
  for (int k0 = 0; k0 < K; k0 += 32) {
    async_wait0();          // own async copies for current tile landed
    __syncthreads();        // every wave's copies visible block-wide
    if (k0 + 32 < K) stage(k0 + 32, buf ^ 1);   // overlap with compute

    HV af[2], bf[4];
    for (int mf = 0; mf < 2; ++mf) {
      const _Float16* ar = &As[buf][(wm * 32 + mf * 16 + lm) * 32];
      af[mf].h[0] = *(const v8h*)(ar + hi * 16);
      af[mf].h[1] = *(const v8h*)(ar + hi * 16 + 8);
    }
    for (int nf = 0; nf < 4; ++nf) {
      const _Float16* br = &Bsh[buf][(wn * 64 + nf * 16 + lm) * 32 + hi * 16];
      bf[nf].h[0] = *(const v8h*)(br);
      bf[nf].h[1] = *(const v8h*)(br + 8);
    }
    for (int mf = 0; mf < 2; ++mf)
      for (int nf = 0; nf < 4; ++nf)
        acc[mf][nf] = wmma_f16(af[mf].v, bf[nf].v, acc[mf][nf]);

    __syncthreads();        // done reading `buf` before it is refilled
    buf ^= 1;
  }
  for (int mf = 0; mf < 2; ++mf)
    for (int nf = 0; nf < 4; ++nf)
      for (int r = 0; r < 8; ++r) {
        int row = mBase + wm * 32 + mf * 16 + r + 8 * hi;
        int col = nBase + wn * 64 + nf * 16 + lm;
        if (col < N) C[(long)row * N + col] = acc[mf][nf][r];
      }
}

// ---------------- depthwise causal conv1d + SiLU ----------------
__global__ void conv_silu_kernel(const float* __restrict__ proj,
                                 const float* __restrict__ cw,
                                 const float* __restrict__ cbias,
                                 float* __restrict__ xconv) {
  long idx = (long)blockIdx.x * blockDim.x + threadIdx.x;  // over B*T*DI
  if (idx >= (long)cB * cT * cDI) return;
  int c = (int)(idx % cDI);
  long bt = idx / cDI;
  int t = (int)(bt % cT);
  long b = bt / cT;
  float acc = cbias[c];
  #pragma unroll
  for (int k = 0; k < 4; ++k) {
    int tt = t - 3 + k;
    if (tt >= 0) acc += cw[c * 4 + k] * proj[(b * cT + tt) * (long)cDP + cDI + c];
  }
  float sig = 1.f / (1.f + __expf(-acc));
  xconv[idx] = acc * sig;
}

// ---------------- B/C slices -> f16 ----------------
__global__ void castbc_kernel(const float* __restrict__ proj,
                              _Float16* __restrict__ Bhp,
                              _Float16* __restrict__ Chp) {
  long idx = (long)blockIdx.x * blockDim.x + threadIdx.x;  // over B*T*DS
  if (idx >= (long)cB * cT * cDS) return;
  int s = (int)(idx % cDS);
  long bt = idx / cDS;
  Bhp[idx] = (_Float16)proj[bt * cDP + 2 * cDI + s];
  Chp[idx] = (_Float16)proj[bt * cDP + 2 * cDI + cDS + s];
}

// ---------------- dt softplus + per-chunk cumsum of dt*A ----------------
__global__ void dtcum_kernel(const float* __restrict__ proj,
                             const float* __restrict__ A_log,
                             const float* __restrict__ dt_bias,
                             float* __restrict__ cum, float* __restrict__ dtv,
                             float* __restrict__ cdec) {
  int wg = blockIdx.x * blockDim.x + threadIdx.x;  // (b*NC + n)*H + h
  if (wg >= cB * cNC * cH) return;
  int h = wg % cH; int bn = wg / cH; int n = bn % cNC; int b = bn / cNC;
  float Ah = -__expf(A_log[h]);
  float run = 0.f;
  for (int t = 0; t < cCS; ++t) {
    float raw = proj[((long)(b * cT + n * cCS + t)) * cDP + 2 * cDI + 2 * cDS + h]
                + dt_bias[h];
    float sp = (raw > 20.f) ? raw : log1pf(__expf(raw));
    dtv[(long)wg * cCS + t] = sp;
    run += sp * Ah;
    cum[(long)wg * cCS + t] = run;
  }
  cdec[wg] = __expf(run);
}

// ---------------- intra-chunk attention-style pass + chunk states ----------------
// one workgroup per (b,n,h): 512 threads = 16 waves; flash-style i/j tiling.
__global__ __launch_bounds__(512) void chunk_kernel(
    const float* __restrict__ xconv, const _Float16* __restrict__ Bh,
    const _Float16* __restrict__ Ch, const float* __restrict__ cum,
    const float* __restrict__ dtv, const float* __restrict__ Dvec,
    float* __restrict__ y, float* __restrict__ cstates) {
  extern __shared__ char smem[];
  _Float16* xdtT = (_Float16*)smem;                 // [64][256]   32 KB (x_dt^T)
  _Float16* BdT  = xdtT + cHD * cCS;                // [128][256]  64 KB (decayed B^T)
  float*    cumS = (float*)(BdT + cDS * cCS);       // [256]        1 KB
  _Float16* pScr = (_Float16*)(cumS + cCS);         // [16][16*32] 16 KB

  const int wg = blockIdx.x;  // (b*NC + n)*H + h
  const int h = wg % cH; const int bn = wg / cH;
  const int n = bn % cNC; const int b = bn / cNC;
  const int tid = threadIdx.x;
  const int lane = tid & 31, wave = tid >> 5;
  const int hi = lane >> 4, lm = lane & 15;
  const long t0 = (long)b * cT + (long)n * cCS;

  for (int t = tid; t < cCS; t += 512) cumS[t] = cum[(long)wg * cCS + t];
  __syncthreads();
  const float cl = cumS[cCS - 1];
  // stage x_dt transposed: xdtT[d][t]
  for (int idx = tid; idx < cHD * cCS; idx += 512) {
    int d = idx & (cHD - 1); int t = idx >> 6;
    float v = xconv[(t0 + t) * cDI + h * cHD + d] * dtv[(long)wg * cCS + t];
    xdtT[d * cCS + t] = (_Float16)v;
  }
  // stage decay-to-end * B transposed: BdT[s][t]
  for (int idx = tid; idx < cDS * cCS; idx += 512) {
    int s = idx & (cDS - 1); int t = idx >> 7;
    float dec = __expf(cl - cumS[t]);
    BdT[s * cCS + t] = (_Float16)((float)Bh[(t0 + t) * cDS + s] * dec);
  }
  __syncthreads();

  // ---- phase 1: y_intra, wave `it` owns 16 rows t in [16*it, 16*it+16) ----
  const int it = wave;
  v8f yacc[4]; for (int f = 0; f < 4; ++f) yacc[f] = vz8();
  const _Float16* Cg = Ch + (t0 + it * 16 + lm) * cDS;  // A-frag rows (contig K=s)

  for (int jt = 0; jt <= it; jt += 2) {
    #pragma unroll
    for (int half = 0; half < 2; ++half) {
      int j = jt + half;
      _Float16* pr = pScr + wave * 512;
      if (j <= it) {
        v8f s8 = vz8();
        #pragma unroll
        for (int ks = 0; ks < cDS; ks += 32) {          // S = C_i * B_j^T
          HV a, bf;
          a.h[0]  = *(const v8h*)(Cg + ks + hi * 16);
          a.h[1]  = *(const v8h*)(Cg + ks + hi * 16 + 8);
          const _Float16* br = Bh + (t0 + j * 16 + lm) * cDS + ks + hi * 16;
          bf.h[0] = *(const v8h*)(br);
          bf.h[1] = *(const v8h*)(br + 8);
          s8 = wmma_f16(a.v, bf.v, s8);
        }
        #pragma unroll
        for (int r = 0; r < 8; ++r) {                   // apply L mask, pack P
          int ti = it * 16 + r + 8 * hi;
          int tj = j * 16 + lm;
          float lv = (ti >= tj) ? __expf(cumS[ti] - cumS[tj]) : 0.f;
          pr[(r + 8 * hi) * 32 + half * 16 + lm] = (_Float16)(s8[r] * lv);
        }
      } else {
        #pragma unroll
        for (int r = 0; r < 8; ++r)
          pr[(r + 8 * hi) * 32 + half * 16 + lm] = (_Float16)0.f;
      }
    }
    // y_i += P(16x32) * x_dt(32 x 64)
    HV ap;
    const _Float16* pr = pScr + wave * 512 + lm * 32;
    ap.h[0] = *(const v8h*)(pr + hi * 16);
    ap.h[1] = *(const v8h*)(pr + hi * 16 + 8);
    #pragma unroll
    for (int nf = 0; nf < 4; ++nf) {
      HV bx;
      const _Float16* br = xdtT + (nf * 16 + lm) * cCS + jt * 16 + hi * 16;
      bx.h[0] = *(const v8h*)(br);
      bx.h[1] = *(const v8h*)(br + 8);
      yacc[nf] = wmma_f16(ap.v, bx.v, yacc[nf]);
    }
  }
  const float Dh = Dvec[h];
  for (int nf = 0; nf < 4; ++nf)
    #pragma unroll
    for (int r = 0; r < 8; ++r) {
      long tg = t0 + it * 16 + r + 8 * hi;
      long off = tg * cDI + h * cHD + nf * 16 + lm;
      y[off] = yacc[nf][r] + Dh * xconv[off];
    }

  // ---- phase 2: chunk_states[d,s] = sum_t decay*x_dt[t,d]*B[t,s] ----
  for (int tidx = wave * 2; tidx < wave * 2 + 2; ++tidx) {
    int mt = tidx >> 3;     // 0..3  (d tiles)
    int nt = tidx & 7;      // 0..7  (s tiles)
    v8f cs8 = vz8();
    #pragma unroll
    for (int kt = 0; kt < cCS; kt += 32) {
      HV a, bf;
      const _Float16* ar = xdtT + (mt * 16 + lm) * cCS + kt;
      a.h[0]  = *(const v8h*)(ar + hi * 16);
      a.h[1]  = *(const v8h*)(ar + hi * 16 + 8);
      const _Float16* br = BdT + (nt * 16 + lm) * cCS + kt + hi * 16;
      bf.h[0] = *(const v8h*)(br);
      bf.h[1] = *(const v8h*)(br + 8);
      cs8 = wmma_f16(a.v, bf.v, cs8);
    }
    long base = (long)wg * cHD * cDS;
    #pragma unroll
    for (int r = 0; r < 8; ++r) {
      int d = mt * 16 + r + 8 * hi;
      cstates[base + d * cDS + nt * 16 + lm] = cs8[r];
    }
  }
}

// ---------------- inter-chunk recurrence (serial over 16 chunks) ----------------
__global__ void scan_kernel(const float* __restrict__ cdec,
                            const float* __restrict__ cstates,
                            float* __restrict__ istates) {
  long idx = (long)blockIdx.x * blockDim.x + threadIdx.x;  // B*H*HD*DS
  if (idx >= (long)cB * cH * cHD * cDS) return;
  int ds = (int)(idx % (cHD * cDS));
  long bh = idx / (cHD * cDS);
  int h = (int)(bh % cH); int b = (int)(bh / cH);
  float s = 0.f;
  for (int n = 0; n < cNC; ++n) {
    long wg = ((long)(b * cNC + n)) * cH + h;
    long off = wg * cHD * cDS + ds;
    istates[off] = s;
    s = cdec[wg] * s + cstates[off];
  }
}

// ---------------- y += decay_from_start * (C * init_state) ----------------
__global__ __launch_bounds__(512) void inter_kernel(
    const _Float16* __restrict__ Ch, const float* __restrict__ istates,
    const float* __restrict__ cum, float* __restrict__ y) {
  __shared__ __align__(32) _Float16 ish[cHD * cDS];  // 16 KB, [d][s]
  __shared__ float cumS[cCS];
  const int wg = blockIdx.x;
  const int h = wg % cH; const int bn = wg / cH;
  const int n = bn % cNC; const int b = bn / cNC;
  const int tid = threadIdx.x;
  const int lane = tid & 31, wave = tid >> 5;
  const int hi = lane >> 4, lm = lane & 15;
  const long t0 = (long)b * cT + (long)n * cCS;

  for (int t = tid; t < cCS; t += 512) cumS[t] = cum[(long)wg * cCS + t];
  long sb = (long)wg * cHD * cDS;
  for (int idx = tid; idx < cHD * cDS; idx += 512)
    ish[idx] = (_Float16)istates[sb + idx];
  __syncthreads();

  const int it = wave;
  const _Float16* Cg = Ch + (t0 + it * 16 + lm) * cDS;
  v8f acc[4]; for (int f = 0; f < 4; ++f) acc[f] = vz8();
  #pragma unroll
  for (int ks = 0; ks < cDS; ks += 32) {
    HV a;
    a.h[0] = *(const v8h*)(Cg + ks + hi * 16);
    a.h[1] = *(const v8h*)(Cg + ks + hi * 16 + 8);
    #pragma unroll
    for (int nf = 0; nf < 4; ++nf) {
      HV bf;
      const _Float16* br = ish + (nf * 16 + lm) * cDS + ks + hi * 16;
      bf.h[0] = *(const v8h*)(br);
      bf.h[1] = *(const v8h*)(br + 8);
      acc[nf] = wmma_f16(a.v, bf.v, acc[nf]);
    }
  }
  for (int nf = 0; nf < 4; ++nf)
    #pragma unroll
    for (int r = 0; r < 8; ++r) {
      int M = r + 8 * hi;
      long tg = t0 + it * 16 + M;
      float dec = __expf(cumS[it * 16 + M]);
      y[tg * cDI + h * cHD + nf * 16 + lm] += acc[nf][r] * dec;
    }
}

// ---------------- RMSNorm * silu(z)  -> f16 for the out-proj ----------------
__global__ __launch_bounds__(256) void normgate_kernel(
    const float* __restrict__ y, const float* __restrict__ proj,
    const float* __restrict__ norm_w, _Float16* __restrict__ ygh) {
  __shared__ float red[256];
  const long row = blockIdx.x;   // B*T rows
  const int tid = threadIdx.x;
  float ss = 0.f;
  for (int c = tid; c < cDI; c += 256) {
    float v = y[row * cDI + c]; ss += v * v;
  }
  red[tid] = ss; __syncthreads();
  for (int st = 128; st > 0; st >>= 1) {
    if (tid < st) red[tid] += red[tid + st];
    __syncthreads();
  }
  const float scale = rsqrtf(red[0] / cDI + 1e-5f);
  for (int c = tid; c < cDI; c += 256) {
    float z = proj[row * cDP + c];
    float sig = 1.f / (1.f + __expf(-z));
    ygh[row * cDI + c] = (_Float16)(y[row * cDI + c] * scale * norm_w[c] * z * sig);
  }
}

// ---------------- host-side orchestration ----------------
extern "C" void kernel_launch(void* const* d_in, const int* in_sizes, int n_in,
                              void* d_out, int out_size, void* d_ws, size_t ws_size,
                              hipStream_t stream) {
  const float* x       = (const float*)d_in[0];
  const float* W_in    = (const float*)d_in[1];
  const float* conv_w  = (const float*)d_in[2];
  const float* conv_b  = (const float*)d_in[3];
  const float* A_log   = (const float*)d_in[4];
  const float* Dvec    = (const float*)d_in[5];
  const float* dt_bias = (const float*)d_in[6];
  const float* norm_w  = (const float*)d_in[7];
  const float* W_out   = (const float*)d_in[8];
  float* out = (float*)d_out;

  char* p = (char*)d_ws;
  auto alloc = [&](long bytes) {
    char* r = p; p += (bytes + 255) & ~255L; return r;
  };
  const long M = (long)cB * cT;                       // 8192 tokens
  _Float16* xh   = (_Float16*)alloc(M * cDM * 2);
  _Float16* Wih  = (_Float16*)alloc((long)cDP * cDM * 2);
  _Float16* Woh  = (_Float16*)alloc((long)cDM * cDI * 2);
  float*    proj = (float*)   alloc(M * cDP * 4);
  float*    xcv  = (float*)   alloc(M * cDI * 4);
  _Float16* Bh   = (_Float16*)alloc(M * cDS * 2);
  _Float16* Chh  = (_Float16*)alloc(M * cDS * 2);
  float*    cum  = (float*)   alloc((long)cB * cNC * cH * cCS * 4);
  float*    dtv  = (float*)   alloc((long)cB * cNC * cH * cCS * 4);
  float*    cdec = (float*)   alloc((long)cB * cNC * cH * 4);
  float*    ybuf = (float*)   alloc(M * cDI * 4);
  float*    cst  = (float*)   alloc((long)cB * cNC * cH * cHD * cDS * 4);
  float*    ist  = (float*)   alloc((long)cB * cNC * cH * cHD * cDS * 4);
  _Float16* ygh  = (_Float16*)alloc(M * cDI * 2);

  // 1) casts to f16
  {
    long n = M * cDM;
    cast16_kernel<<<(unsigned)((n + 255) / 256), 256, 0, stream>>>(x, xh, n);
    n = (long)cDP * cDM;
    cast16_kernel<<<(unsigned)((n + 255) / 256), 256, 0, stream>>>(W_in, Wih, n);
    n = (long)cDM * cDI;
    cast16_kernel<<<(unsigned)((n + 255) / 256), 256, 0, stream>>>(W_out, Woh, n);
  }
  // 2) in-projection: proj = x @ W_in^T   (8192 x 4384, K=1024)
  gemm_f16_kernel<<<dim3((cDP + 127) / 128, (unsigned)(M / 128)), 256, 0, stream>>>(
      xh, Wih, proj, (int)M, cDP, cDM);
  // 3) depthwise causal conv + silu
  {
    long n = M * cDI;
    conv_silu_kernel<<<(unsigned)((n + 255) / 256), 256, 0, stream>>>(
        proj, conv_w, conv_b, xcv);
  }
  // 4) B/C -> f16
  {
    long n = M * cDS;
    castbc_kernel<<<(unsigned)((n + 255) / 256), 256, 0, stream>>>(proj, Bh, Chh);
  }
  // 5) dt softplus + chunk cumsum
  dtcum_kernel<<<(cB * cNC * cH + 255) / 256, 256, 0, stream>>>(
      proj, A_log, dt_bias, cum, dtv, cdec);
  // 6) intra-chunk (WMMA) + chunk states (WMMA); 113 KB dynamic LDS
  {
    const unsigned smem = (cHD * cCS + cDS * cCS + 16 * 512) * 2 + cCS * 4;
    chunk_kernel<<<cB * cNC * cH, 512, smem, stream>>>(
        xcv, Bh, Chh, cum, dtv, Dvec, ybuf, cst);
  }
  // 7) inter-chunk scan
  {
    long n = (long)cB * cH * cHD * cDS;
    scan_kernel<<<(unsigned)((n + 255) / 256), 256, 0, stream>>>(cdec, cst, ist);
  }
  // 8) y += decay * C @ init_state (WMMA)
  inter_kernel<<<cB * cNC * cH, 512, 0, stream>>>(Chh, ist, cum, ybuf);
  // 9) RMSNorm + gate -> f16
  normgate_kernel<<<(unsigned)M, 256, 0, stream>>>(ybuf, proj, norm_w, ygh);
  // 10) out-projection: out = yg @ W_out^T (8192 x 1024, K=2048)
  gemm_f16_kernel<<<dim3((cDM + 127) / 128, (unsigned)(M / 128)), 256, 0, stream>>>(
      ygh, Woh, out, (int)M, cDM, cDI);
}
